// GFAEncoder_48842368090245
// MI455X (gfx1250) — matrix operand order repaired and checked by reading, not compiled, not run on hardware
//
#include <hip/hip_runtime.h>
#include <hip/hip_bf16.h>
#include <math.h>

typedef _Float16 f16;
typedef _Float16 v8h  __attribute__((ext_vector_type(8)));
typedef _Float16 v16h __attribute__((ext_vector_type(16)));
typedef float    v8f  __attribute__((ext_vector_type(8)));

#define B_    16
#define C_    384
#define FS_   56
#define NH_   8
#define HD_   48
#define HP_   28
#define NTOK_ 784            // 28*28
#define M1_   (B_*NTOK_)     // 12544
#define HW_   (FS_*FS_)      // 3136
#define M2_   (B_*HW_)       // 50176
#define QKVW_ (3*C_)         // 1152

// ---------------------------------------------------------------- converts
__global__ void cvt_f32_f16(const float* __restrict__ s, f16* __restrict__ d, int n) {
  int i = blockIdx.x * 256 + threadIdx.x;
  if (i < n) d[i] = (f16)s[i];
}

// ct_w [c, d, 2, 2] (f32) -> ctw_h [pq][d][c] (f16) so each pq slab is W[N=d, K=c]
__global__ void cvt_ctw(const float* __restrict__ s, f16* __restrict__ d) {
  int i = blockIdx.x * 256 + threadIdx.x;
  if (i >= 4 * C_ * C_) return;
  int pq = i / (C_ * C_);
  int r  = i % (C_ * C_);
  int dc = r / C_;
  int c  = r % C_;
  d[i] = (f16)s[(c * C_ + dc) * 4 + pq];
}

// ------------------------------------------- depthwise conv + residual + avgpool
__global__ void dwpool_kernel(const float* __restrict__ x, const float* __restrict__ w,
                              const float* __restrict__ bias, float* __restrict__ y) {
  int idx = blockIdx.x * 256 + threadIdx.x;
  if (idx >= B_ * C_ * HP_ * HP_) return;
  int j = idx % HP_;
  int i = (idx / HP_) % HP_;
  int c = (idx / (HP_ * HP_)) % C_;
  int b = idx / (HP_ * HP_ * C_);
  const float* xc = x + (size_t)(b * C_ + c) * HW_;
  float wk[9];
#pragma unroll
  for (int t = 0; t < 9; ++t) wk[t] = w[c * 9 + t];
  float bb = bias[c];
  float acc = 0.f;
#pragma unroll
  for (int p = 0; p < 2; ++p)
#pragma unroll
    for (int q = 0; q < 2; ++q) {
      int h0 = 2 * i + p, w0 = 2 * j + q;
      float s = bb;
#pragma unroll
      for (int r = 0; r < 3; ++r) {
        int hh = h0 - 1 + r;
        if (hh < 0 || hh >= FS_) continue;
#pragma unroll
        for (int t = 0; t < 3; ++t) {
          int ww = w0 - 1 + t;
          if (ww < 0 || ww >= FS_) continue;
          s += xc[hh * FS_ + ww] * wk[r * 3 + t];
        }
      }
      acc += s + xc[h0 * FS_ + w0];
    }
  y[(size_t)(b * C_ + c) * NTOK_ + i * HP_ + j] = acc * 0.25f;
}

// --------------------------------------------------------- LayerNorm #1 (over C)
__global__ void ln1_kernel(const float* __restrict__ y, const float* __restrict__ g,
                           const float* __restrict__ be, f16* __restrict__ t) {
  int m = blockIdx.x;            // token in [0, 12544)
  int b = m / NTOK_;
  int n = m % NTOK_;
  int tid = threadIdx.x;         // 128
  float v[3], s = 0.f, ss = 0.f;
#pragma unroll
  for (int r = 0; r < 3; ++r) {
    int c = tid + r * 128;
    float val = y[(size_t)(b * C_ + c) * NTOK_ + n];
    v[r] = val; s += val; ss += val * val;
  }
  __shared__ float r1[128], r2[128];
  r1[tid] = s; r2[tid] = ss;
  __syncthreads();
  for (int o = 64; o > 0; o >>= 1) {
    if (tid < o) { r1[tid] += r1[tid + o]; r2[tid] += r2[tid + o]; }
    __syncthreads();
  }
  float mean = r1[0] * (1.f / C_);
  float var  = r2[0] * (1.f / C_) - mean * mean;
  float rstd = rsqrtf(var + 1e-6f);
#pragma unroll
  for (int r = 0; r < 3; ++r) {
    int c = tid + r * 128;
    t[(size_t)m * C_ + c] = (f16)((v[r] - mean) * rstd * g[c] + be[c]);
  }
}

// ----------------------- LayerNorm #2: gathers ConvTranspose output, LN over C
__global__ void ln2_kernel(const f16* __restrict__ cth, const float* __restrict__ g,
                           const float* __restrict__ be, f16* __restrict__ u) {
  int r0 = blockIdx.x;           // [0, 50176)  row = (b, h0, w0)
  int b  = r0 / HW_;
  int hw = r0 % HW_;
  int h0 = hw / FS_, w0 = hw % FS_;
  int i = h0 >> 1, p = h0 & 1, j = w0 >> 1, q = w0 & 1;
  int pq = p * 2 + q;
  const f16* src = cth + (size_t)(pq * M1_ + b * NTOK_ + i * HP_ + j) * C_;
  int tid = threadIdx.x;         // 128
  float v[3], s = 0.f, ss = 0.f;
#pragma unroll
  for (int r = 0; r < 3; ++r) {
    int c = tid + r * 128;
    float val = (float)src[c];
    v[r] = val; s += val; ss += val * val;
  }
  __shared__ float r1[128], r2[128];
  r1[tid] = s; r2[tid] = ss;
  __syncthreads();
  for (int o = 64; o > 0; o >>= 1) {
    if (tid < o) { r1[tid] += r1[tid + o]; r2[tid] += r2[tid + o]; }
    __syncthreads();
  }
  float mean = r1[0] * (1.f / C_);
  float var  = r2[0] * (1.f / C_) - mean * mean;
  float rstd = rsqrtf(var + 1e-6f);
#pragma unroll
  for (int r = 0; r < 3; ++r) {
    int c = tid + r * 128;
    u[(size_t)r0 * C_ + c] = (f16)((v[r] - mean) * rstd * g[c] + be[c]);
  }
}

// --------------------------------------------------------------- WMMA GEMM
// D[m,n] = sum_k A[m,k] * W[n,k]  (+ bias[n]), f16 in / f32 accumulate.
// Block = 128 threads = 4 waves; each wave owns a 32(M) x 64(N) tile:
// 2 A fragments share 4 B fragments -> 8 WMMAs per 12 b128 loads per K-step.
// K/N compile-time: secondary load addresses become immediate offsets from just
// two moving base pointers (no 64-bit adds, no spills in the hot loop).
// MODE: 0 f32 store | 1 f16 store of 2*val | 2 f16 store | 3 f16 GELU | 4 f32
//       residual store transposed into [B,C,56,56] (xres added).
template <int K, int N, int MODE>
__global__ __launch_bounds__(128, 1) void wmma_gemm_kernel(
    const f16* __restrict__ A, const f16* __restrict__ W,
    const float* __restrict__ bias, float* __restrict__ oF, f16* __restrict__ oH,
    const float* __restrict__ xres) {
  int lane = threadIdx.x & 31;
  int wave = threadIdx.x >> 5;
  int m0 = (blockIdx.y * 4 + wave) * 32;
  int n0 = blockIdx.x * 64;
  int lr = lane & 15;
  int hi = lane >> 4;                 // 0 or 1 (half-wave)
  v8f acc[8];
#pragma unroll
  for (int t = 0; t < 8; ++t) acc[t] = {};
  // A fragment: lane holds row m0+lr, 16 contiguous K at ka = hi*16
  const f16* pa = A + (size_t)(m0 + lr) * K + hi * 16;
  // B fragment: lane holds col n, 16 contiguous K at kb = hi*16
  const f16* pw = W + (size_t)(n0 + lr) * K + hi * 16;
  for (int kt = 0; kt < K; kt += 32) {
    union { v16h v; v8h h[2]; } af0, af1;
    af0.h[0] = *(const v8h*)(pa);
    af0.h[1] = *(const v8h*)(pa + 8);
    af1.h[0] = *(const v8h*)(pa + 16 * K);
    af1.h[1] = *(const v8h*)(pa + 16 * K + 8);
    v16h b0 = *(const v16h*)(pw);
    v16h b1 = *(const v16h*)(pw + 16 * K);
    v16h b2 = *(const v16h*)(pw + 32 * K);
    v16h b3 = *(const v16h*)(pw + 48 * K);
    __builtin_prefetch(pa + 32, 0, 1);
    __builtin_prefetch(pw + 32, 0, 1);
    acc[0] = __builtin_amdgcn_wmma_f32_16x16x32_f16(false, af0.v, false, b0, (short)0, acc[0], false, false);
    acc[1] = __builtin_amdgcn_wmma_f32_16x16x32_f16(false, af0.v, false, b1, (short)0, acc[1], false, false);
    acc[2] = __builtin_amdgcn_wmma_f32_16x16x32_f16(false, af0.v, false, b2, (short)0, acc[2], false, false);
    acc[3] = __builtin_amdgcn_wmma_f32_16x16x32_f16(false, af0.v, false, b3, (short)0, acc[3], false, false);
    acc[4] = __builtin_amdgcn_wmma_f32_16x16x32_f16(false, af1.v, false, b0, (short)0, acc[4], false, false);
    acc[5] = __builtin_amdgcn_wmma_f32_16x16x32_f16(false, af1.v, false, b1, (short)0, acc[5], false, false);
    acc[6] = __builtin_amdgcn_wmma_f32_16x16x32_f16(false, af1.v, false, b2, (short)0, acc[6], false, false);
    acc[7] = __builtin_amdgcn_wmma_f32_16x16x32_f16(false, af1.v, false, b3, (short)0, acc[7], false, false);
    pa += 32; pw += 32;
  }
#pragma unroll
  for (int mt = 0; mt < 2; ++mt) {
#pragma unroll
    for (int nt = 0; nt < 4; ++nt) {
      v8f a = acc[mt * 4 + nt];
      int n = n0 + nt * 16 + lr;
      float bs = bias[n];
#pragma unroll
      for (int i2 = 0; i2 < 8; ++i2) {
        int m = m0 + mt * 16 + i2 + hi * 8; // C/D: VGPR i -> M=i (lanes 0-15), M=8+i (16-31)
        float val = a[i2] + bs;
        if constexpr (MODE == 0) {
          oF[(size_t)m * N + n] = val;
        } else if constexpr (MODE == 1) {
          oH[(size_t)m * N + n] = (f16)(2.f * val);
        } else if constexpr (MODE == 2) {
          oH[(size_t)m * N + n] = (f16)val;
        } else if constexpr (MODE == 3) {
          float gl = 0.5f * val * (1.f + erff(val * 0.70710678118f));
          oH[(size_t)m * N + n] = (f16)gl;
        } else {
          int bb = m / HW_;
          int hw = m % HW_;
          size_t di = (size_t)(bb * C_ + n) * HW_ + hw;
          oF[di] = xres[di] + val;
        }
      }
    }
  }
}

// ------------------------------------------------------------- attention
// One block per (b,h). q,k L2-normalized over tokens; attn 48x48; softmax; o=attn@v.
#define CH_   56
#define QSTR  57
__global__ void attn_kernel(const float* __restrict__ qkv, const unsigned char* __restrict__ mask,
                            const float* __restrict__ temp, f16* __restrict__ o2) {
  int bh = blockIdx.x;
  int b = bh >> 3;
  int h = bh & 7;
  int tid = threadIdx.x;           // 256
  const float* base = qkv + (size_t)b * NTOK_ * QKVW_;
  int qo = h * HD_;
  int ko = C_ + h * HD_;
  int vo = 2 * C_ + h * HD_;
  __shared__ float sQ[HD_ * QSTR];
  __shared__ float sK[HD_ * QSTR];
  __shared__ float sA[HD_ * HD_];
  __shared__ float sIq[HD_], sIk[HD_];

  // 1/||row|| for q and k (96 rows of 784)
  for (int r = tid; r < 2 * HD_; r += 256) {
    int isK = r >= HD_;
    int d = isK ? r - HD_ : r;
    int off = (isK ? ko : qo) + d;
    float ss = 0.f;
    for (int n = 0; n < NTOK_; ++n) {
      float v = base[(size_t)n * QKVW_ + off];
      ss += v * v;
    }
    float inv = 1.f / fmaxf(sqrtf(ss), 1e-12f);
    if (isK) sIk[d] = inv; else sIq[d] = inv;
  }

  // attn[d][e] = q[d].k[e], LDS-chunked over tokens (2304 pairs = 9 per thread)
  float accp[9];
#pragma unroll
  for (int s = 0; s < 9; ++s) accp[s] = 0.f;
  for (int ch = 0; ch < NTOK_; ch += CH_) {
    __syncthreads();
    for (int l = tid; l < HD_ * CH_; l += 256) {
      int d = l / CH_;
      int cc = l % CH_;
      sQ[d * QSTR + cc] = base[(size_t)(ch + cc) * QKVW_ + qo + d];
      sK[d * QSTR + cc] = base[(size_t)(ch + cc) * QKVW_ + ko + d];
    }
    __syncthreads();
#pragma unroll
    for (int s = 0; s < 9; ++s) {
      int pair = s * 256 + tid;
      int d = pair / HD_;
      int e = pair % HD_;
      const float* qr = &sQ[d * QSTR];
      const float* kr = &sK[e * QSTR];
      float sum = accp[s];
      for (int cc = 0; cc < CH_; ++cc) sum += qr[cc] * kr[cc];
      accp[s] = sum;
    }
  }
  __syncthreads();
  float tp = temp[h];
  const unsigned char* mrow = mask + (size_t)bh * HD_ * HD_;
#pragma unroll
  for (int s = 0; s < 9; ++s) {
    int pair = s * 256 + tid;
    int d = pair / HD_;
    int e = pair % HD_;
    float a = accp[s] * tp * sIq[d] * sIk[e];
    if (mrow[pair]) a -= 1e12f;
    sA[pair] = a;
  }
  __syncthreads();
  // softmax over e per row d
  for (int d = tid; d < HD_; d += 256) {
    float* row = &sA[d * HD_];
    float mx = row[0];
    for (int e = 1; e < HD_; ++e) mx = fmaxf(mx, row[e]);
    float sm = 0.f;
    for (int e = 0; e < HD_; ++e) { float ex = __expf(row[e] - mx); row[e] = ex; sm += ex; }
    float inv = 1.f / sm;
    for (int e = 0; e < HD_; ++e) row[e] *= inv;
  }
  __syncthreads();
  // o[d][n] = sum_e attn[d][e]*v[e][n]; stored as [B*N, C] f16 for the proj GEMM
  for (int l = tid; l < HD_ * NTOK_; l += 256) {
    int d = l / NTOK_;
    int n = l % NTOK_;
    const float* vr = base + (size_t)n * QKVW_ + vo;
    const float* ar = &sA[d * HD_];
    float sum = 0.f;
#pragma unroll 8
    for (int e = 0; e < HD_; ++e) sum += ar[e] * vr[e];
    o2[(size_t)(b * NTOK_ + n) * C_ + h * HD_ + d] = (f16)sum;
  }
}

// ================================================================= host
extern "C" void kernel_launch(void* const* d_in, const int* in_sizes, int n_in,
                              void* d_out, int out_size, void* d_ws, size_t ws_size,
                              hipStream_t stream) {
  (void)in_sizes; (void)n_in; (void)out_size; (void)ws_size;
  const float*         x      = (const float*)d_in[0];
  const unsigned char* mask   = (const unsigned char*)d_in[1];
  const float*         dw_w   = (const float*)d_in[2];
  const float*         dw_b   = (const float*)d_in[3];
  const float*         n1g    = (const float*)d_in[4];
  const float*         n1b    = (const float*)d_in[5];
  const float*         temp   = (const float*)d_in[6];
  const float*         qkv_w  = (const float*)d_in[7];
  const float*         qkv_b  = (const float*)d_in[8];
  const float*         proj_w = (const float*)d_in[9];
  const float*         proj_b = (const float*)d_in[10];
  const float*         ct_w   = (const float*)d_in[11];
  const float*         ct_b   = (const float*)d_in[12];
  const float*         n2g    = (const float*)d_in[13];
  const float*         n2b    = (const float*)d_in[14];
  const float*         pw1_w  = (const float*)d_in[15];
  const float*         pw1_b  = (const float*)d_in[16];
  const float*         pw2_w  = (const float*)d_in[17];
  const float*         pw2_b  = (const float*)d_in[18];
  float* out = (float*)d_out;

  char* ws = (char*)d_ws;
  size_t off = 0;
  auto alloc = [&](size_t bytes) -> void* {
    void* p = ws + off;
    off += (bytes + 255) & ~(size_t)255;
    return p;
  };
  f16*   qkvw_h  = (f16*)  alloc((size_t)QKVW_ * C_ * 2);
  f16*   projw_h = (f16*)  alloc((size_t)C_ * C_ * 2);
  f16*   ctw_h   = (f16*)  alloc((size_t)4 * C_ * C_ * 2);
  f16*   pw1w_h  = (f16*)  alloc((size_t)4 * C_ * C_ * 2);
  f16*   pw2w_h  = (f16*)  alloc((size_t)4 * C_ * C_ * 2);
  float* y       = (float*)alloc((size_t)M1_ * C_ * 4);
  f16*   t_h     = (f16*)  alloc((size_t)M1_ * C_ * 2);
  float* qkv_f   = (float*)alloc((size_t)M1_ * QKVW_ * 4);
  f16*   o2_h    = (f16*)  alloc((size_t)M1_ * C_ * 2);
  f16*   proj2_h = (f16*)  alloc((size_t)M1_ * C_ * 2);
  f16*   ct_h    = (f16*)  alloc((size_t)4 * M1_ * C_ * 2);
  f16*   u_h     = (f16*)  alloc((size_t)M2_ * C_ * 2);
  f16*   g_h     = (f16*)  alloc((size_t)M2_ * 4 * C_ * 2);

  dim3 blk256(256), blk128(128);
  auto g1 = [](int n) { return dim3((n + 255) / 256); };

  // weight conversion
  cvt_f32_f16<<<g1(QKVW_ * C_), blk256, 0, stream>>>(qkv_w, qkvw_h, QKVW_ * C_);
  cvt_f32_f16<<<g1(C_ * C_), blk256, 0, stream>>>(proj_w, projw_h, C_ * C_);
  cvt_f32_f16<<<g1(4 * C_ * C_), blk256, 0, stream>>>(pw1_w, pw1w_h, 4 * C_ * C_);
  cvt_f32_f16<<<g1(4 * C_ * C_), blk256, 0, stream>>>(pw2_w, pw2w_h, 4 * C_ * C_);
  cvt_ctw<<<g1(4 * C_ * C_), blk256, 0, stream>>>(ct_w, ctw_h);

  // dwconv + residual + avgpool -> y [B,C,28,28]
  dwpool_kernel<<<g1(B_ * C_ * HP_ * HP_), blk256, 0, stream>>>(x, dw_w, dw_b, y);

  // LN1 -> t_h [12544, 384] f16
  ln1_kernel<<<dim3(M1_), blk128, 0, stream>>>(y, n1g, n1b, t_h);

  // qkv GEMM: [12544,384] x [1152,384]^T -> f32
  wmma_gemm_kernel<C_, QKVW_, 0><<<dim3(QKVW_ / 64, M1_ / 128), blk128, 0, stream>>>(
      t_h, qkvw_h, qkv_b, qkv_f, nullptr, nullptr);

  // channel attention per (b,h) -> o2_h [12544, 384] f16
  attn_kernel<<<dim3(B_ * NH_), blk256, 0, stream>>>(qkv_f, mask, temp, o2_h);

  // proj GEMM (doubled) -> proj2_h f16
  wmma_gemm_kernel<C_, C_, 1><<<dim3(C_ / 64, M1_ / 128), blk128, 0, stream>>>(
      o2_h, projw_h, proj_b, nullptr, proj2_h, nullptr);

  // ConvTranspose 2x2 = 4 GEMMs -> ct_h[pq] f16 (ct_b fused)
  for (int pq = 0; pq < 4; ++pq) {
    wmma_gemm_kernel<C_, C_, 2><<<dim3(C_ / 64, M1_ / 128), blk128, 0, stream>>>(
        proj2_h, ctw_h + (size_t)pq * C_ * C_, ct_b,
        nullptr, ct_h + (size_t)pq * M1_ * C_, nullptr);
  }

  // LN2 (scatter-gather of the 2x2 upsample) -> u_h [50176, 384] f16
  ln2_kernel<<<dim3(M2_), blk128, 0, stream>>>(ct_h, n2g, n2b, u_h);

  // MLP: pw1 + GELU -> g_h [50176, 1536] f16
  wmma_gemm_kernel<C_, 4 * C_, 3><<<dim3(4 * C_ / 64, M2_ / 128), blk128, 0, stream>>>(
      u_h, pw1w_h, pw1_b, nullptr, g_h, nullptr);

  // pw2 + transpose to NCHW + input residual -> d_out
  wmma_gemm_kernel<4 * C_, C_, 4><<<dim3(C_ / 64, M2_ / 128), blk128, 0, stream>>>(
      g_h, pw2w_h, pw2_b, out, nullptr, x);
}
